// MechanismHypergraphModel_20109036880618
// MI455X (gfx1250) — compile-verified, matrix operand order checked by reading, not run on hardware
//
#include <hip/hip_runtime.h>
#include <hip/hip_bf16.h>

typedef __attribute__((ext_vector_type(16))) _Float16 v16h;
typedef __attribute__((ext_vector_type(8)))  _Float16 v8h;
typedef __attribute__((ext_vector_type(8)))  float    v8f;

// Problem constants (match reference)
constexpr int NUM_GENES = 50000;
constexpr int NUM_EDGES = 5000;
constexpr int NNZ       = 1600000;
constexpr int P         = 500;
constexpr int PPAD      = 512;   // p padded for 16-tiles (zero-filled)
constexpr int D         = 128;
constexpr int B         = 64;
constexpr int M         = 200;
constexpr int MPAD      = 224;   // K padded to 7*32 (zero-filled)
constexpr int R         = B * P; // 32000 rows for MLP GEMM

// ---------------- workspace layout (bytes) ----------------
constexpr size_t al512(size_t x) { return (x + 511) & ~size_t(511); }
constexpr size_t DV_OFF   = 0;                                           // [NUM_GENES] f32 (deg -> dv)
constexpr size_t DE_OFF   = al512(DV_OFF   + size_t(NUM_GENES) * 4);     // [NUM_EDGES] f32 (deg -> de)
constexpr size_t HX_OFF   = al512(DE_OFF   + size_t(NUM_EDGES) * 4);     // [NUM_EDGES*D] f32
constexpr size_t XP_OFF   = al512(HX_OFF   + size_t(NUM_EDGES) * D * 4); // [NUM_GENES*D] f32
constexpr size_t XHT_OFF  = al512(XP_OFF   + size_t(NUM_GENES) * D * 4); // [B*D*MPAD] f16  (K-major B operand)
constexpr size_t MSKT_OFF = al512(XHT_OFF  + size_t(B) * D * MPAD * 2);  // [B*PPAD*MPAD] f16 (masks^T, K-major)
constexpr size_t CTXH_OFF = al512(MSKT_OFF + size_t(B) * PPAD * MPAD * 2);// [B*D] f16
constexpr size_t W1T_OFF  = al512(CTXH_OFF + size_t(B) * D * 2);         // [D*2D] f16 (W1 transposed, K-major)
constexpr size_t ICNT_OFF = al512(W1T_OFF  + size_t(D) * 2 * D * 2);     // [B*P] f32 (1/clip(count,1))
constexpr size_t REPF_OFF = al512(ICNT_OFF + size_t(R) * 4);             // [B*P*D] f32
constexpr size_t REPH_OFF = al512(REPF_OFF + size_t(R) * D * 4);         // [B*P*D] f16
constexpr size_t H_OFF    = al512(REPH_OFF + size_t(R) * D * 2);         // [B*P*D] f32 (tanh MLP out)
constexpr size_t SC_OFF   = al512(H_OFF    + size_t(R) * D * 4);         // [R] f32 scores
constexpr size_t WSM_OFF  = al512(SC_OFF   + size_t(R) * 4);             // [R] f32 softmax w
constexpr size_t Z0_OFF   = al512(WSM_OFF  + size_t(R) * 4);             // [B*D] f32
constexpr size_t ZERO_N   = (XP_OFF + size_t(NUM_GENES) * D * 4) / 4;    // floats to zero (dv..Xp)

// Build a 16-half fragment from two contiguous 8-half groups
__device__ __forceinline__ v16h frag16(v8h a, v8h b) {
  return __builtin_shufflevector(a, b, 0, 1, 2, 3, 4, 5, 6, 7, 8, 9, 10, 11, 12, 13, 14, 15);
}

// ---------------- kernels ----------------

__global__ void zero_kernel(float* p, size_t n) {
  size_t i = (size_t)blockIdx.x * blockDim.x + threadIdx.x;
  size_t stride = (size_t)gridDim.x * blockDim.x;
  for (; i < n; i += stride) p[i] = 0.f;
}

__global__ void degrees_kernel(const int* __restrict__ rows, const int* __restrict__ cols,
                               const float* __restrict__ vals, float* dv, float* de) {
  int i = blockIdx.x * blockDim.x + threadIdx.x;
  if (i >= NNZ) return;
  float v = vals[i];
  atomicAdd(dv + rows[i], v);
  atomicAdd(de + cols[i], v);
}

__global__ void finalize_deg_kernel(float* dv, float* de) {
  int i = blockIdx.x * blockDim.x + threadIdx.x;
  if (i < NUM_GENES) dv[i] = rsqrtf(dv[i] + 1e-6f);
  if (i < NUM_EDGES) de[i] = 1.f / (de[i] + 1e-6f);
}

// HX[col] += val * (gene_embed[row] * dv[row]); one wave per nnz, lane owns float4
__global__ void pass1_kernel(const int* __restrict__ rows, const int* __restrict__ cols,
                             const float* __restrict__ vals, const float* __restrict__ E,
                             const float* __restrict__ dv, float* __restrict__ HX) {
  int i = blockIdx.x * 8 + (threadIdx.x >> 5);
  int lane = threadIdx.x & 31;
  if (i >= NNZ) return;
  int r = rows[i], c = cols[i];
  float s = vals[i] * dv[r];
  const float4 x = *reinterpret_cast<const float4*>(E + (size_t)r * D + lane * 4);
  float* dst = HX + (size_t)c * D + lane * 4;
  atomicAdd(dst + 0, s * x.x);
  atomicAdd(dst + 1, s * x.y);
  atomicAdd(dst + 2, s * x.z);
  atomicAdd(dst + 3, s * x.w);
}

__global__ void scale_hx_kernel(float* HX, const float* __restrict__ de) {
  int i = blockIdx.x * blockDim.x + threadIdx.x;
  if (i < NUM_EDGES * D) HX[i] *= de[i / D];
}

// Xp[row] += val * HX[col]
__global__ void pass2_kernel(const int* __restrict__ rows, const int* __restrict__ cols,
                             const float* __restrict__ vals, const float* __restrict__ HX,
                             float* __restrict__ Xp) {
  int i = blockIdx.x * 8 + (threadIdx.x >> 5);
  int lane = threadIdx.x & 31;
  if (i >= NNZ) return;
  int r = rows[i], c = cols[i];
  float s = vals[i];
  const float4 x = *reinterpret_cast<const float4*>(HX + (size_t)c * D + lane * 4);
  float* dst = Xp + (size_t)r * D + lane * 4;
  atomicAdd(dst + 0, s * x.x);
  atomicAdd(dst + 1, s * x.y);
  atomicAdd(dst + 2, s * x.z);
  atomicAdd(dst + 3, s * x.w);
}

// XhT[b][n][m] = E_prop[gid[b,m]][n] (f16), zero-padded m in [M, MPAD)
__global__ void gather_xT_kernel(const int* __restrict__ gene_ids, const float* __restrict__ Xp,
                                 const float* __restrict__ dv, _Float16* __restrict__ XhT) {
  int i = blockIdx.x * blockDim.x + threadIdx.x;  // over B*D*MPAD, m innermost
  if (i >= B * D * MPAD) return;
  int m = i % MPAD, n = (i / MPAD) % D, b = i / (MPAD * D);
  _Float16 v = (_Float16)0.f;
  if (m < M) {
    int g = gene_ids[b * M + m];
    v = (_Float16)(Xp[(size_t)g * D + n] * dv[g]);
  }
  XhT[i] = v;
}

// maskT[b][p][m] = gene_pathway[gid[b,m], p] (f16), zero-padded; LDS tile transpose.
// grid = B * 16(pTiles of 32) * 7(mTiles of 32), block 256
__global__ void maskT_kernel(const int* __restrict__ gene_ids, const float* __restrict__ gp,
                             _Float16* __restrict__ maskT) {
  __shared__ _Float16 tile[32][33];
  __shared__ int gid_s[32];
  int bid = blockIdx.x;
  int mt = bid % 7, pt = (bid / 7) % 16, b = bid / (7 * 16);
  int t = threadIdx.x;
  if (t < 32) {
    int m = mt * 32 + t;
    gid_s[t] = (m < M) ? gene_ids[b * M + m] : -1;
  }
  __syncthreads();
  int pl = t & 31, mr = t >> 5;
#pragma unroll
  for (int it = 0; it < 4; ++it) {
    int mloc = mr + 8 * it;
    int p = pt * 32 + pl;
    int g = gid_s[mloc];
    float v = (g >= 0 && p < P) ? gp[(size_t)g * P + p] : 0.f;  // coalesced over p
    tile[mloc][pl] = (_Float16)v;
  }
  __syncthreads();
  int ml = t & 31, pr = t >> 5;
#pragma unroll
  for (int it = 0; it < 4; ++it) {
    int ploc = pr + 8 * it;
    size_t row = (size_t)b * PPAD + pt * 32 + ploc;
    maskT[row * MPAD + mt * 32 + ml] = tile[ml][ploc];          // coalesced over m
  }
}

// W1T[n][k] = W1[k][n] (f16); ctxh[b][:] = treatment_embed[context_ids[b]] (f16)
__global__ void prep_kernel(const float* __restrict__ W1, const int* __restrict__ ctx_ids,
                            const float* __restrict__ treat, _Float16* __restrict__ W1T,
                            _Float16* __restrict__ ctxh) {
  int i = blockIdx.x * blockDim.x + threadIdx.x;
  if (i < 2 * D * D) {
    int n = i % D, k = i / D;
    W1T[(size_t)n * 256 + k] = (_Float16)W1[i];
  }
  if (i < B * D) {
    int b = i >> 7;
    ctxh[i] = (_Float16)treat[(size_t)ctx_ids[b] * D + (i & 127)];
  }
}

// inv_counts[b,p] = 1 / max(1, sum_m gene_pathway[gid[b,m], p])
__global__ void counts_kernel(const int* __restrict__ gene_ids,
                              const float* __restrict__ gp, float* __restrict__ inv_cnt) {
  __shared__ int gid_s[M];
  int b = blockIdx.x;
  for (int i = threadIdx.x; i < M; i += blockDim.x) gid_s[i] = gene_ids[b * M + i];
  __syncthreads();
  int p = threadIdx.x;
  if (p >= P) return;
  float c = 0.f;
  for (int m = 0; m < M; ++m) c += gp[(size_t)gid_s[m] * P + p];
  inv_cnt[b * P + p] = 1.f / fmaxf(c, 1.f);
}

// GEMM1: rep[b] = (masks[b]^T @ X[b]) * inv_cnt. grid = B*32 (b, p-tile), 8 waves = 8 d-tiles.
// All operands pre-padded & K-contiguous -> 4x global_load_b128 + 1x wmma per k-step.
__global__ __launch_bounds__(256) void gemm1_wmma(const _Float16* __restrict__ maskT,
                                                  const _Float16* __restrict__ XhT,
                                                  const float* __restrict__ inv_cnt,
                                                  float* __restrict__ repf,
                                                  _Float16* __restrict__ reph) {
  int b = blockIdx.x >> 5;
  int pt = blockIdx.x & 31;
  int dt = threadIdx.x >> 5;
  int lane = threadIdx.x & 31;
  int hi = lane >> 4, lo = lane & 15;
  int p = pt * 16 + lo;
  int n = dt * 16 + lo;
  const _Float16* arow = maskT + ((size_t)b * PPAD + p) * MPAD;
  const _Float16* brow = XhT + ((size_t)b * D + n) * MPAD;
  v8f acc = {};
#pragma unroll
  for (int kt = 0; kt < 7; ++kt) {
    int k0 = kt * 32 + hi * 8;
    v16h a = frag16(*reinterpret_cast<const v8h*>(arow + k0),
                    *reinterpret_cast<const v8h*>(arow + k0 + 16));
    v16h bf = frag16(*reinterpret_cast<const v8h*>(brow + k0),
                     *reinterpret_cast<const v8h*>(brow + k0 + 16));
    acc = __builtin_amdgcn_wmma_f32_16x16x32_f16(false, a, false, bf, (short)0, acc, false, false);
  }
#pragma unroll
  for (int r = 0; r < 8; ++r) {
    int pr = pt * 16 + r + 8 * hi;
    if (pr < P) {
      float v = acc[r] * inv_cnt[b * P + pr];
      size_t idx = (size_t)(b * P + pr) * D + n;
      repf[idx] = v;
      reph[idx] = (_Float16)v;
    }
  }
}

// GEMM2: h = tanh([rep|ctx] @ W1 + b1). 2000 row-tiles, K=256, 8 n-tiles (waves).
__global__ __launch_bounds__(256) void gemm2_wmma(const _Float16* __restrict__ reph,
                                                  const _Float16* __restrict__ ctxh,
                                                  const _Float16* __restrict__ W1T,
                                                  const float* __restrict__ b1,
                                                  float* __restrict__ hout) {
  int rowTile = blockIdx.x;  // 0..1999
  int nt = threadIdx.x >> 5;
  int lane = threadIdx.x & 31;
  int hi = lane >> 4, lo = lane & 15;
  int rowA = rowTile * 16 + lo;
  int bA = rowA / P;
  int n = nt * 16 + lo;
  const _Float16* arow0 = reph + (size_t)rowA * D;  // k in [0,128)
  const _Float16* arow1 = ctxh + (size_t)bA * D;    // k-128 in [0,128)
  const _Float16* brow = W1T + (size_t)n * 256;
  v8f acc = {};
#pragma unroll
  for (int kt = 0; kt < 8; ++kt) {
    int k0 = kt * 32 + hi * 8;
    const _Float16* ar = (kt < 4) ? (arow0 + k0) : (arow1 + (k0 - 128));
    v16h a = frag16(*reinterpret_cast<const v8h*>(ar),
                    *reinterpret_cast<const v8h*>(ar + 16));
    v16h bf = frag16(*reinterpret_cast<const v8h*>(brow + k0),
                     *reinterpret_cast<const v8h*>(brow + k0 + 16));
    acc = __builtin_amdgcn_wmma_f32_16x16x32_f16(false, a, false, bf, (short)0, acc, false, false);
  }
  float bn = b1[n];
#pragma unroll
  for (int r = 0; r < 8; ++r) {
    int row = rowTile * 16 + r + 8 * hi;
    hout[(size_t)row * D + n] = tanhf(acc[r] + bn);
  }
}

// scores[row] = h[row,:] . W2 + b2 ; one wave per row
__global__ void scores_kernel(const float* __restrict__ h, const float* __restrict__ W2,
                              const float* __restrict__ b2, float* __restrict__ scores) {
  int wave = threadIdx.x >> 5, lane = threadIdx.x & 31;
  int row = blockIdx.x * 8 + wave;
  float s = 0.f;
#pragma unroll
  for (int i = 0; i < 4; ++i) {
    int d = lane + 32 * i;
    s += h[(size_t)row * D + d] * W2[d];
  }
#pragma unroll
  for (int off = 16; off > 0; off >>= 1) s += __shfl_down(s, off, 32);
  if (lane == 0) scores[row] = s + b2[0];
}

__global__ void softmax_kernel(const float* __restrict__ scores, float* __restrict__ w) {
  __shared__ float sd[512];
  int b = blockIdx.x, tid = threadIdx.x;
  float v = (tid < P) ? scores[b * P + tid] : -3.4e38f;
  sd[tid] = v;
  __syncthreads();
  for (int s = 256; s > 0; s >>= 1) {
    if (tid < s) sd[tid] = fmaxf(sd[tid], sd[tid + s]);
    __syncthreads();
  }
  float mx = sd[0];
  __syncthreads();
  float e = (tid < P) ? __expf(v - mx) : 0.f;
  sd[tid] = e;
  __syncthreads();
  for (int s = 256; s > 0; s >>= 1) {
    if (tid < s) sd[tid] += sd[tid + s];
    __syncthreads();
  }
  float inv = 1.f / sd[0];
  if (tid < P) w[b * P + tid] = e * inv;
}

// z0[b,d] = sum_p w[b,p] * rep[b,p,d]
__global__ void z0_kernel(const float* __restrict__ w, const float* __restrict__ repf,
                          float* __restrict__ z0) {
  __shared__ float ws[P];
  int b = blockIdx.x, d = threadIdx.x;
  for (int i = threadIdx.x; i < P; i += blockDim.x) ws[i] = w[b * P + i];
  __syncthreads();
  float acc = 0.f;
  for (int p = 0; p < P; ++p) acc = fmaf(ws[p], repf[(size_t)(b * P + p) * D + d], acc);
  z0[b * D + d] = acc;
}

// z = z0 @ latent_W + latent_b ; risk = z . risk_W + risk_b
// out[0..B) = risk ; out[B + b*D + j] = z
__global__ void final_kernel(const float* __restrict__ z0, const float* __restrict__ LW,
                             const float* __restrict__ Lb, const float* __restrict__ rW,
                             const float* __restrict__ rb, float* __restrict__ out) {
  __shared__ float z0s[D];
  __shared__ float prod[D];
  int b = blockIdx.x, j = threadIdx.x;
  z0s[j] = z0[b * D + j];
  __syncthreads();
  float zj = Lb[j];
  for (int k = 0; k < D; ++k) zj = fmaf(z0s[k], LW[(size_t)k * D + j], zj);
  out[B + (size_t)b * D + j] = zj;
  prod[j] = zj * rW[j];
  __syncthreads();
  for (int s = 64; s > 0; s >>= 1) {
    if (j < s) prod[j] += prod[j + s];
    __syncthreads();
  }
  if (j == 0) out[b] = prod[0] + rb[0];
}

// ---------------- host launcher ----------------
extern "C" void kernel_launch(void* const* d_in, const int* in_sizes, int n_in,
                              void* d_out, int out_size, void* d_ws, size_t ws_size,
                              hipStream_t stream) {
  const int*   gene_ids    = (const int*)d_in[0];
  const int*   context_ids = (const int*)d_in[1];
  const int*   H_rows      = (const int*)d_in[2];
  const int*   H_cols      = (const int*)d_in[3];
  const float* H_vals      = (const float*)d_in[4];
  const float* gene_embed  = (const float*)d_in[5];
  const float* treat_embed = (const float*)d_in[6];
  const float* gene_path   = (const float*)d_in[7];
  const float* W1          = (const float*)d_in[8];
  const float* b1          = (const float*)d_in[9];
  const float* W2          = (const float*)d_in[10];
  const float* b2          = (const float*)d_in[11];
  const float* latent_W    = (const float*)d_in[12];
  const float* latent_b    = (const float*)d_in[13];
  const float* risk_W      = (const float*)d_in[14];
  const float* risk_b      = (const float*)d_in[15];
  float* out = (float*)d_out;

  char* ws = (char*)d_ws;
  float*    dv      = (float*)(ws + DV_OFF);
  float*    de      = (float*)(ws + DE_OFF);
  float*    HX      = (float*)(ws + HX_OFF);
  float*    Xp      = (float*)(ws + XP_OFF);
  _Float16* XhT     = (_Float16*)(ws + XHT_OFF);
  _Float16* maskT   = (_Float16*)(ws + MSKT_OFF);
  _Float16* ctxh    = (_Float16*)(ws + CTXH_OFF);
  _Float16* W1T     = (_Float16*)(ws + W1T_OFF);
  float*    inv_cnt = (float*)(ws + ICNT_OFF);
  float*    repf    = (float*)(ws + REPF_OFF);
  _Float16* reph    = (_Float16*)(ws + REPH_OFF);
  float*    hbuf    = (float*)(ws + H_OFF);
  float*    scores  = (float*)(ws + SC_OFF);
  float*    wsm     = (float*)(ws + WSM_OFF);
  float*    z0      = (float*)(ws + Z0_OFF);

  // 1) zero accumulators (dv, de, HX, Xp are one contiguous prefix)
  zero_kernel<<<4096, 256, 0, stream>>>(dv, ZERO_N);
  // 2) degrees + normalization factors
  degrees_kernel<<<(NNZ + 255) / 256, 256, 0, stream>>>(H_rows, H_cols, H_vals, dv, de);
  finalize_deg_kernel<<<(NUM_GENES + 255) / 256, 256, 0, stream>>>(dv, de);
  // 3) HX = H^T (dv * E)   (scatter, wave per nnz)
  pass1_kernel<<<(NNZ + 7) / 8, 256, 0, stream>>>(H_rows, H_cols, H_vals, gene_embed, dv, HX);
  // 4) HX *= de
  scale_hx_kernel<<<(NUM_EDGES * D + 255) / 256, 256, 0, stream>>>(HX, de);
  // 5) Xp = H HX (scatter)
  pass2_kernel<<<(NNZ + 7) / 8, 256, 0, stream>>>(H_rows, H_cols, H_vals, HX, Xp);
  // 6) WMMA operand staging: XhT (K-major, padded), masks^T (padded), W1T/ctx f16
  gather_xT_kernel<<<(B * D * MPAD + 255) / 256, 256, 0, stream>>>(gene_ids, Xp, dv, XhT);
  maskT_kernel<<<B * 16 * 7, 256, 0, stream>>>(gene_ids, gene_path, maskT);
  prep_kernel<<<(2 * D * D + 255) / 256, 256, 0, stream>>>(W1, context_ids, treat_embed, W1T, ctxh);
  // 7) pathway member counts
  counts_kernel<<<B, 512, 0, stream>>>(gene_ids, gene_path, inv_cnt);
  // 8) WMMA GEMM1: rep = (masks^T @ X) / counts
  gemm1_wmma<<<B * 32, 256, 0, stream>>>(maskT, XhT, inv_cnt, repf, reph);
  // 9) WMMA GEMM2: h = tanh([rep|ctx] @ W1 + b1)
  gemm2_wmma<<<R / 16, 256, 0, stream>>>(reph, ctxh, W1T, b1, hbuf);
  // 10) scores = h @ W2 + b2
  scores_kernel<<<R / 8, 256, 0, stream>>>(hbuf, W2, b2, scores);
  // 11) softmax over pathways
  softmax_kernel<<<B, 512, 0, stream>>>(scores, wsm);
  // 12) z0 = sum_p w * rep
  z0_kernel<<<B, D, 0, stream>>>(wsm, repf, z0);
  // 13) z = z0 @ latent_W + latent_b ; risk = z . risk_W + risk_b
  final_kernel<<<B, D, 0, stream>>>(z0, latent_W, latent_b, risk_W, risk_b, out);
  (void)in_sizes; (void)n_in; (void)out_size; (void)ws_size;
}
